// PhonemeLangModelv2_54623394071118
// MI455X (gfx1250) — matrix operand order repaired and checked by reading, not compiled, not run on hardware
//
#include <hip/hip_runtime.h>
#include <hip/hip_bf16.h>
#include <cstdint>
#include <cstddef>

typedef __attribute__((ext_vector_type(16))) _Float16 v16h;
typedef __attribute__((ext_vector_type(8)))  _Float16 v8h;
typedef __attribute__((ext_vector_type(8)))  float    v8f;

namespace {

constexpr int Bb = 256;   // batch
constexpr int Ss = 500;   // encoder seq len
constexpr int Hh = 128;   // hidden
constexpr int Gg = 512;   // 4*H gates
constexpr int Vv = 46;    // vocab
constexpr int Tt = 500;   // decoder steps

__device__ __forceinline__ float sigm(float x) { return 1.f / (1.f + __expf(-x)); }

// Optimization barrier: an opaque zero. Added to loop-invariant pointers inside
// the persistent scan so LICM cannot hoist the weight loads out of the 500-step
// loop (which previously spilled ~1000 VGPRs of hoisted B-fragments to scratch).
__device__ __forceinline__ int opaque_zero() {
  int x = 0;
  asm volatile("" : "+v"(x));
  return x;
}

// CDNA5 async copy: global memory -> LDS without touching VGPR data paths.
// lds_byte_off is the per-lane LDS byte address (dynamic-LDS base offset 0).
// Tracked by ASYNCcnt; wait with s_wait_asynccnt.
__device__ __forceinline__ void async_copy_b128(uint32_t lds_byte_off, const void* gaddr) {
  asm volatile("global_load_async_to_lds_b128 %0, %1, off"
               :
               : "v"(lds_byte_off), "v"(gaddr)
               : "memory");
}

__device__ __forceinline__ void wait_asynccnt0() {
  asm volatile("s_wait_asynccnt 0x0" ::: "memory");
}

__device__ __forceinline__ v8f vzero8() {
  v8f r;
#pragma unroll
  for (int i = 0; i < 8; i++) r[i] = 0.f;
  return r;
}

// A-fragment (16x32 f16, M=batch rows, K=feature cols) for v_wmma_f32_16x16x32_f16.
// Per ISA 7.12.2: lanes 0-15 hold M=lane, K in {k0..k0+7, k0+16..k0+23};
//                 lanes 16-31 hold M=lane-16, K in {k0+8..k0+15, k0+24..k0+31}.
__device__ __forceinline__ v16h load_a_frag(const _Float16* base, int stride, int m0, int k0) {
  const int lane = threadIdx.x & 31;
  const _Float16* p =
      base + (size_t)(m0 + (lane & 15)) * stride + k0 + ((lane & 16) ? 8 : 0);
  v8h c0 = *(const v8h*)(p);
  v8h c1 = *(const v8h*)(p + 16);
  v16h r;
#pragma unroll
  for (int i = 0; i < 8; i++) { r[i] = c0[i]; r[i + 8] = c1[i]; }
  return r;
}

// B-fragment from pre-swizzled buffer: frag (kt,nt) is 32 lanes x 16 halfs (1KB).
__device__ __forceinline__ v16h load_b_frag(const _Float16* frags, int kt, int nt, int NT) {
  const int lane = threadIdx.x & 31;
  return *(const v16h*)(frags + ((size_t)(kt * NT + nt) * 512) + lane * 16);
}

__device__ __forceinline__ v8f wmma16(v16h a, v16h b, v8f c) {
  return __builtin_amdgcn_wmma_f32_16x16x32_f16(false, a, false, b, (short)0, c, false, false);
}

}  // namespace

// ---------------------------------------------------------------------------
// Pre-swizzle W (N x K row-major, we need B = W.T as KxN tiles) into WMMA
// B-fragment layout: lanes0-15 -> N=nt*16+lane, K=kt*32+{0..7,16..23}; lanes16-31 -> K+8.
// ---------------------------------------------------------------------------
__global__ void plm_pack_frags(const float* __restrict__ W, _Float16* __restrict__ dst,
                               int K, int Npad, int N) {
  const int NT = Npad >> 4;
  const int total = (K >> 5) * NT * 512;
  for (int idx = blockIdx.x * blockDim.x + threadIdx.x; idx < total;
       idx += gridDim.x * blockDim.x) {
    const int kk = idx & 15;
    const int lane = (idx >> 4) & 31;
    const int f = idx >> 9;
    const int kt = f / NT, nt = f % NT;
    const int n = nt * 16 + (lane & 15);
    const int k = kt * 32 + ((lane & 16) >> 1) + ((kk & 8) << 1) + (kk & 7);
    const float w = (n < N) ? W[(size_t)n * K + k] : 0.f;
    dst[idx] = (_Float16)w;
  }
}

// ---------------------------------------------------------------------------
// Encoder layer 0 (one block per direction). Persistent scan over S steps.
// LDS: Whh B-fragments (128KB, async-staged) + h hi/lo f16 (128KB).
// Gates = (h_hi + h_lo) @ Whh^T  (WMMA, f32 acc)  +  x_t * Wih  +  b.
// ---------------------------------------------------------------------------
__global__ __launch_bounds__(1024) void plm_enc_l0(
    const int* __restrict__ x, const float* __restrict__ h0, const float* __restrict__ c0,
    const float* __restrict__ wih_all, const float* __restrict__ bias_all,
    const _Float16* __restrict__ whh_frags_all,
    _Float16* __restrict__ cur_hi, _Float16* __restrict__ cur_lo,
    float* __restrict__ h_fin, float* __restrict__ c_fin) {
  const int dir = blockIdx.x;
  const int dl = dir;  // 2*layer + dir, layer = 0
  const float* wih_g = wih_all + dir * Gg;
  const float* bias_g = bias_all + dir * Gg;

  extern __shared__ char smem_raw[];
  _Float16* whhL = (_Float16*)smem_raw;            // 4*32 frags = 65536 halfs (128KB)
  _Float16* h_hi = whhL + 4 * 32 * 512;            // 256*128
  _Float16* h_lo = h_hi + Bb * Hh;                 // 256*128

  // Async-stage the recurrent-weight fragments into LDS (one-time): the
  // fragment region starts at dynamic-LDS offset 0.
  {
    const uint4* src = (const uint4*)(whh_frags_all + (size_t)dir * (4 * 32 * 512));
    for (int e = threadIdx.x; e < (4 * 32 * 512 * 2) / 16; e += blockDim.x)
      async_copy_b128((uint32_t)e * 16u, src + e);
  }
  for (int e = threadIdx.x; e < Bb * Hh; e += blockDim.x) {
    float h = h0[(size_t)dl * Bb * Hh + e];
    _Float16 hi = (_Float16)h;
    h_hi[e] = hi;
    h_lo[e] = (_Float16)(h - (float)hi);
  }

  const int wave = threadIdx.x >> 5;
  const int lane = threadIdx.x & 31;
  const int moff = (lane & 16) ? 8 : 0;
  const int ncol = lane & 15;

  v8f ct[4];
#pragma unroll
  for (int i = 0; i < 4; i++) {
    const int q = wave * 4 + i;                       // 128 patches / 32 waves
    const int m0 = (q >> 3) << 4, j0 = (q & 7) << 4;
#pragma unroll
    for (int r = 0; r < 8; r++) {
      const int m = m0 + r + moff;
      ct[i][r] = c0[(size_t)dl * Bb * Hh + (size_t)m * Hh + j0 + ncol];
    }
  }
  wait_asynccnt0();
  __syncthreads();

  for (int step = 0; step < Ss; ++step) {
    const int t = dir ? (Ss - 1 - step) : step;
    const int oz = opaque_zero();          // defeat LICM on weight loads
    const _Float16* whh = whhL + oz;
    const float* wih = wih_g + oz;
    const float* bias = bias_g + oz;
    v8f hnew[4];
#pragma unroll
    for (int i = 0; i < 4; i++) {
      const int q = wave * 4 + i;
      const int m0 = (q >> 3) << 4, j0 = (q & 7) << 4;
      const int ntb = j0 >> 4;
      v8f a0 = vzero8(), a1 = vzero8(), a2 = vzero8(), a3 = vzero8();
#pragma unroll
      for (int kt = 0; kt < 4; ++kt) {
        const v16h ahi = load_a_frag(h_hi, Hh, m0, kt * 32);
        const v16h alo = load_a_frag(h_lo, Hh, m0, kt * 32);
        v16h bf;
        bf = load_b_frag(whh, kt, 0 * 8 + ntb, 32);
        a0 = wmma16(ahi, bf, a0); a0 = wmma16(alo, bf, a0);
        bf = load_b_frag(whh, kt, 1 * 8 + ntb, 32);
        a1 = wmma16(ahi, bf, a1); a1 = wmma16(alo, bf, a1);
        bf = load_b_frag(whh, kt, 2 * 8 + ntb, 32);
        a2 = wmma16(ahi, bf, a2); a2 = wmma16(alo, bf, a2);
        bf = load_b_frag(whh, kt, 3 * 8 + ntb, 32);
        a3 = wmma16(ahi, bf, a3); a3 = wmma16(alo, bf, a3);
      }
      const int n = j0 + ncol;
      const float wi = wih[n], wf = wih[Hh + n], wg = wih[2 * Hh + n], wo = wih[3 * Hh + n];
      const float bi = bias[n], bf_ = bias[Hh + n], bg = bias[2 * Hh + n], bo = bias[3 * Hh + n];
#pragma unroll
      for (int r = 0; r < 8; r++) {
        const int m = m0 + r + moff;
        const float xv = (float)x[(size_t)m * Ss + t];
        const float pi = a0[r] + xv * wi + bi;
        const float pf = a1[r] + xv * wf + bf_;
        const float pg = a2[r] + xv * wg + bg;
        const float po = a3[r] + xv * wo + bo;
        const float cn = sigm(pf) * ct[i][r] + sigm(pi) * tanhf(pg);
        ct[i][r] = cn;
        hnew[i][r] = sigm(po) * tanhf(cn);
      }
    }
    __syncthreads();  // all reads of old h complete
#pragma unroll
    for (int i = 0; i < 4; i++) {
      const int q = wave * 4 + i;
      const int m0 = (q >> 3) << 4, j0 = (q & 7) << 4;
      const int n = j0 + ncol;
#pragma unroll
      for (int r = 0; r < 8; r++) {
        const int m = m0 + r + moff;
        const float h = hnew[i][r];
        const _Float16 hi = (_Float16)h;
        const _Float16 lo = (_Float16)(h - (float)hi);
        h_hi[(size_t)m * Hh + n] = hi;
        h_lo[(size_t)m * Hh + n] = lo;
        const size_t gix = ((size_t)t * Bb + m) * (2 * Hh) + dir * Hh + n;
        cur_hi[gix] = hi;   // layer-1 input sequence (concat along features)
        cur_lo[gix] = lo;
      }
    }
    __syncthreads();
  }

  for (int e = threadIdx.x; e < Bb * Hh; e += blockDim.x)
    h_fin[(size_t)dl * Bb * Hh + e] = (float)h_hi[e] + (float)h_lo[e];
#pragma unroll
  for (int i = 0; i < 4; i++) {
    const int q = wave * 4 + i;
    const int m0 = (q >> 3) << 4, j0 = (q & 7) << 4;
    const int n = j0 + ncol;
#pragma unroll
    for (int r = 0; r < 8; r++) {
      const int m = m0 + r + moff;
      c_fin[(size_t)dl * Bb * Hh + (size_t)m * Hh + n] = ct[i][r];
    }
  }
}

// ---------------------------------------------------------------------------
// Encoder layer 1 (one block per direction): input GEMM (K=256, fragments from
// L2, next-step rows prefetched) + recurrent GEMM (K=128, Whh in LDS).
// ---------------------------------------------------------------------------
__global__ __launch_bounds__(1024) void plm_enc_l1(
    const float* __restrict__ h0, const float* __restrict__ c0,
    const float* __restrict__ bias_all,
    const _Float16* __restrict__ wih_frags_all, const _Float16* __restrict__ whh_frags_all,
    const _Float16* __restrict__ cur_hi, const _Float16* __restrict__ cur_lo,
    float* __restrict__ h_fin, float* __restrict__ c_fin) {
  const int dir = blockIdx.x;
  const int dl = 2 + dir;
  const _Float16* wih_g = wih_frags_all + (size_t)dir * (8 * 32 * 512);
  const float* bias_g = bias_all + dir * Gg;

  extern __shared__ char smem_raw[];
  _Float16* whhL = (_Float16*)smem_raw;            // 128KB, async-staged
  _Float16* h_hi = whhL + 4 * 32 * 512;
  _Float16* h_lo = h_hi + Bb * Hh;

  {
    const uint4* src = (const uint4*)(whh_frags_all + (size_t)dir * (4 * 32 * 512));
    for (int e = threadIdx.x; e < (4 * 32 * 512 * 2) / 16; e += blockDim.x)
      async_copy_b128((uint32_t)e * 16u, src + e);
  }
  for (int e = threadIdx.x; e < Bb * Hh; e += blockDim.x) {
    float h = h0[(size_t)dl * Bb * Hh + e];
    _Float16 hi = (_Float16)h;
    h_hi[e] = hi;
    h_lo[e] = (_Float16)(h - (float)hi);
  }

  const int wave = threadIdx.x >> 5;
  const int lane = threadIdx.x & 31;
  const int moff = (lane & 16) ? 8 : 0;
  const int ncol = lane & 15;

  v8f ct[4];
#pragma unroll
  for (int i = 0; i < 4; i++) {
    const int q = wave * 4 + i;
    const int m0 = (q >> 3) << 4, j0 = (q & 7) << 4;
#pragma unroll
    for (int r = 0; r < 8; r++) {
      const int m = m0 + r + moff;
      ct[i][r] = c0[(size_t)dl * Bb * Hh + (size_t)m * Hh + j0 + ncol];
    }
  }
  wait_asynccnt0();
  __syncthreads();

  for (int step = 0; step < Ss; ++step) {
    const int t = dir ? (Ss - 1 - step) : step;
    const int oz = opaque_zero();
    const _Float16* whh = whhL + oz;
    const _Float16* wih = wih_g + oz;
    const float* bias = bias_g + oz;
    const _Float16* Ah = cur_hi + (size_t)t * Bb * (2 * Hh);
    const _Float16* Al = cur_lo + (size_t)t * Bb * (2 * Hh);

    // Prefetch next step's input rows into L2/L0 (global_prefetch_b8).
    if (step + 1 < Ss) {
      const int tn = dir ? (Ss - 2 - step) : (step + 1);
      const _Float16* An = cur_hi + (size_t)tn * Bb * (2 * Hh);
      const _Float16* Aln = cur_lo + (size_t)tn * Bb * (2 * Hh);
#pragma unroll
      for (int i = 0; i < 4; i++) {
        const int q = wave * 4 + i;
        const int m0 = (q >> 3) << 4;
        const size_t off = (size_t)(m0 + (lane & 15)) * (2 * Hh) + ((lane & 16) ? 128 : 0);
        __builtin_prefetch(An + off, 0, 3);
        __builtin_prefetch(Aln + off, 0, 3);
      }
    }

    v8f hnew[4];
#pragma unroll
    for (int i = 0; i < 4; i++) {
      const int q = wave * 4 + i;
      const int m0 = (q >> 3) << 4, j0 = (q & 7) << 4;
      const int ntb = j0 >> 4;
      v8f a0 = vzero8(), a1 = vzero8(), a2 = vzero8(), a3 = vzero8();
#pragma unroll
      for (int kt = 0; kt < 8; ++kt) {  // input K = 256
        const v16h ahi = load_a_frag(Ah, 2 * Hh, m0, kt * 32);
        const v16h alo = load_a_frag(Al, 2 * Hh, m0, kt * 32);
        v16h bf;
        bf = load_b_frag(wih, kt, 0 * 8 + ntb, 32);
        a0 = wmma16(ahi, bf, a0); a0 = wmma16(alo, bf, a0);
        bf = load_b_frag(wih, kt, 1 * 8 + ntb, 32);
        a1 = wmma16(ahi, bf, a1); a1 = wmma16(alo, bf, a1);
        bf = load_b_frag(wih, kt, 2 * 8 + ntb, 32);
        a2 = wmma16(ahi, bf, a2); a2 = wmma16(alo, bf, a2);
        bf = load_b_frag(wih, kt, 3 * 8 + ntb, 32);
        a3 = wmma16(ahi, bf, a3); a3 = wmma16(alo, bf, a3);
      }
#pragma unroll
      for (int kt = 0; kt < 4; ++kt) {  // recurrent K = 128 (LDS fragments)
        const v16h ahi = load_a_frag(h_hi, Hh, m0, kt * 32);
        const v16h alo = load_a_frag(h_lo, Hh, m0, kt * 32);
        v16h bf;
        bf = load_b_frag(whh, kt, 0 * 8 + ntb, 32);
        a0 = wmma16(ahi, bf, a0); a0 = wmma16(alo, bf, a0);
        bf = load_b_frag(whh, kt, 1 * 8 + ntb, 32);
        a1 = wmma16(ahi, bf, a1); a1 = wmma16(alo, bf, a1);
        bf = load_b_frag(whh, kt, 2 * 8 + ntb, 32);
        a2 = wmma16(ahi, bf, a2); a2 = wmma16(alo, bf, a2);
        bf = load_b_frag(whh, kt, 3 * 8 + ntb, 32);
        a3 = wmma16(ahi, bf, a3); a3 = wmma16(alo, bf, a3);
      }
      const int n = j0 + ncol;
      const float bi = bias[n], bf_ = bias[Hh + n], bg = bias[2 * Hh + n], bo = bias[3 * Hh + n];
#pragma unroll
      for (int r = 0; r < 8; r++) {
        const float pi = a0[r] + bi;
        const float pf = a1[r] + bf_;
        const float pg = a2[r] + bg;
        const float po = a3[r] + bo;
        const float cn = sigm(pf) * ct[i][r] + sigm(pi) * tanhf(pg);
        ct[i][r] = cn;
        hnew[i][r] = sigm(po) * tanhf(cn);
      }
    }
    __syncthreads();
#pragma unroll
    for (int i = 0; i < 4; i++) {
      const int q = wave * 4 + i;
      const int m0 = (q >> 3) << 4, j0 = (q & 7) << 4;
      const int n = j0 + ncol;
#pragma unroll
      for (int r = 0; r < 8; r++) {
        const int m = m0 + r + moff;
        const float h = hnew[i][r];
        const _Float16 hi = (_Float16)h;
        h_hi[(size_t)m * Hh + n] = hi;
        h_lo[(size_t)m * Hh + n] = (_Float16)(h - (float)hi);
      }
    }
    __syncthreads();
  }

  for (int e = threadIdx.x; e < Bb * Hh; e += blockDim.x)
    h_fin[(size_t)dl * Bb * Hh + e] = (float)h_hi[e] + (float)h_lo[e];
#pragma unroll
  for (int i = 0; i < 4; i++) {
    const int q = wave * 4 + i;
    const int m0 = (q >> 3) << 4, j0 = (q & 7) << 4;
    const int n = j0 + ncol;
#pragma unroll
    for (int r = 0; r < 8; r++) {
      const int m = m0 + r + moff;
      c_fin[(size_t)dl * Bb * Hh + (size_t)m * Hh + n] = ct[i][r];
    }
  }
}

// ---------------------------------------------------------------------------
// Decoder: single persistent workgroup. Per step: both L0 cells, both L1
// cells, logits GEMM (N padded 46->48), argmax + blank-flag masking, output.
// c state lives in L2 (updated in place in c_fin, rewritten by the encoder
// every call) to keep VGPR pressure low; h state (f16) lives in LDS.
// ---------------------------------------------------------------------------
__global__ __launch_bounds__(1024) void plm_decoder(
    const float* __restrict__ h_fin, float* __restrict__ c_fin,
    const float* __restrict__ wih0_all, const float* __restrict__ b0_all,
    const float* __restrict__ b1_all,
    const _Float16* __restrict__ whh0_frags_all, const _Float16* __restrict__ whh1_frags_all,
    const _Float16* __restrict__ wih1_frags_all, const _Float16* __restrict__ lin_frags,
    const float* __restrict__ lin_b, float* __restrict__ out) {
  extern __shared__ char smem_raw[];
  _Float16* hA = (_Float16*)smem_raw;                                 // 4 x (256*128) f16
  float* logits = (float*)(smem_raw + (size_t)4 * Bb * Hh * 2);       // 256*48 f32
  float* yv = (float*)(smem_raw + (size_t)4 * Bb * Hh * 2 + (size_t)Bb * 48 * 4);
  int* flg = (int*)(smem_raw + (size_t)4 * Bb * Hh * 2 + (size_t)Bb * 48 * 4 + Bb * 4);

  const int wave = threadIdx.x >> 5;
  const int lane = threadIdx.x & 31;
  const int moff = (lane & 16) ? 8 : 0;
  const int ncol = lane & 15;

  for (int e = threadIdx.x; e < 4 * Bb * Hh; e += blockDim.x)
    hA[e] = (_Float16)h_fin[e];
  if (threadIdx.x < Bb) { yv[threadIdx.x] = 1.0f; flg[threadIdx.x] = 0; }  // MASK_IDX=1
  __syncthreads();

  for (int t = 0; t < Tt; ++t) {
    const int oz = opaque_zero();
    const _Float16* whh0f = whh0_frags_all + oz;
    const _Float16* whh1f = whh1_frags_all + oz;
    const _Float16* wih1f = wih1_frags_all + oz;
    const _Float16* linf = lin_frags + oz;
    const float* wih0 = wih0_all + oz;
    const float* b0p = b0_all + oz;
    const float* b1p = b1_all + oz;
    float* cfin = c_fin + oz;
    v8f hnew[8];

    // ---------------- layer 0 ----------------
#pragma unroll
    for (int i = 0; i < 8; i++) {
      const int q = wave * 8 + i;  // [0,256): dir = q>>7
      const int d = q >> 7;
      const int w = q & 127;
      const int m0 = (w >> 3) << 4, j0 = (w & 7) << 4;
      const int ntb = j0 >> 4;
      const _Float16* hsrc = hA + (size_t)d * Bb * Hh;
      const _Float16* whh = whh0f + (size_t)d * (4 * 32 * 512);
      v8f a0 = vzero8(), a1 = vzero8(), a2 = vzero8(), a3 = vzero8();
#pragma unroll
      for (int kt = 0; kt < 4; ++kt) {
        const v16h af = load_a_frag(hsrc, Hh, m0, kt * 32);
        a0 = wmma16(af, load_b_frag(whh, kt, 0 * 8 + ntb, 32), a0);
        a1 = wmma16(af, load_b_frag(whh, kt, 1 * 8 + ntb, 32), a1);
        a2 = wmma16(af, load_b_frag(whh, kt, 2 * 8 + ntb, 32), a2);
        a3 = wmma16(af, load_b_frag(whh, kt, 3 * 8 + ntb, 32), a3);
      }
      const int n = j0 + ncol;
      const float* wih = wih0 + d * Gg;
      const float* bias = b0p + d * Gg;
      float* cbase = cfin + (size_t)d * Bb * Hh;  // layer-0 state, in place
      const float wi = wih[n], wf = wih[Hh + n], wg = wih[2 * Hh + n], wo = wih[3 * Hh + n];
      const float bi = bias[n], bf_ = bias[Hh + n], bg = bias[2 * Hh + n], bo = bias[3 * Hh + n];
#pragma unroll
      for (int r = 0; r < 8; r++) {
        const int m = m0 + r + moff;
        const float xv = yv[m];
        const float pi = a0[r] + xv * wi + bi;
        const float pf = a1[r] + xv * wf + bf_;
        const float pg = a2[r] + xv * wg + bg;
        const float po = a3[r] + xv * wo + bo;
        const float cn = sigm(pf) * cbase[(size_t)m * Hh + n] + sigm(pi) * tanhf(pg);
        cbase[(size_t)m * Hh + n] = cn;
        hnew[i][r] = sigm(po) * tanhf(cn);
      }
    }
    __syncthreads();
#pragma unroll
    for (int i = 0; i < 8; i++) {
      const int q = wave * 8 + i;
      const int d = q >> 7;
      const int w = q & 127;
      const int m0 = (w >> 3) << 4, j0 = (w & 7) << 4;
      _Float16* hdst = hA + (size_t)d * Bb * Hh;
#pragma unroll
      for (int r = 0; r < 8; r++)
        hdst[(size_t)(m0 + r + moff) * Hh + j0 + ncol] = (_Float16)hnew[i][r];
    }
    __syncthreads();

    // ---------------- layer 1 ----------------
#pragma unroll
    for (int i = 0; i < 8; i++) {
      const int q = wave * 8 + i;
      const int d = q >> 7;
      const int w = q & 127;
      const int m0 = (w >> 3) << 4, j0 = (w & 7) << 4;
      const int ntb = j0 >> 4;
      const _Float16* whh = whh1f + (size_t)d * (4 * 32 * 512);
      const _Float16* wih = wih1f + (size_t)d * (8 * 32 * 512);
      v8f a0 = vzero8(), a1 = vzero8(), a2 = vzero8(), a3 = vzero8();
#pragma unroll
      for (int kt = 0; kt < 8; ++kt) {  // input = concat(h0f, h0b)
        const _Float16* src = (kt < 4) ? hA : (hA + (size_t)Bb * Hh);
        const int k0 = (kt < 4) ? kt * 32 : (kt - 4) * 32;
        const v16h af = load_a_frag(src, Hh, m0, k0);
        a0 = wmma16(af, load_b_frag(wih, kt, 0 * 8 + ntb, 32), a0);
        a1 = wmma16(af, load_b_frag(wih, kt, 1 * 8 + ntb, 32), a1);
        a2 = wmma16(af, load_b_frag(wih, kt, 2 * 8 + ntb, 32), a2);
        a3 = wmma16(af, load_b_frag(wih, kt, 3 * 8 + ntb, 32), a3);
      }
      const _Float16* hsrc = hA + (size_t)(2 + d) * Bb * Hh;
#pragma unroll
      for (int kt = 0; kt < 4; ++kt) {
        const v16h af = load_a_frag(hsrc, Hh, m0, kt * 32);
        a0 = wmma16(af, load_b_frag(whh, kt, 0 * 8 + ntb, 32), a0);
        a1 = wmma16(af, load_b_frag(whh, kt, 1 * 8 + ntb, 32), a1);
        a2 = wmma16(af, load_b_frag(whh, kt, 2 * 8 + ntb, 32), a2);
        a3 = wmma16(af, load_b_frag(whh, kt, 3 * 8 + ntb, 32), a3);
      }
      const int n = j0 + ncol;
      const float* bias = b1p + d * Gg;
      float* cbase = cfin + (size_t)(2 + d) * Bb * Hh;  // layer-1 state, in place
      const float bi = bias[n], bf_ = bias[Hh + n], bg = bias[2 * Hh + n], bo = bias[3 * Hh + n];
#pragma unroll
      for (int r = 0; r < 8; r++) {
        const int m = m0 + r + moff;
        const float pi = a0[r] + bi;
        const float pf = a1[r] + bf_;
        const float pg = a2[r] + bg;
        const float po = a3[r] + bo;
        const float cn = sigm(pf) * cbase[(size_t)m * Hh + n] + sigm(pi) * tanhf(pg);
        cbase[(size_t)m * Hh + n] = cn;
        hnew[i][r] = sigm(po) * tanhf(cn);
      }
    }
    __syncthreads();
#pragma unroll
    for (int i = 0; i < 8; i++) {
      const int q = wave * 8 + i;
      const int d = q >> 7;
      const int w = q & 127;
      const int m0 = (w >> 3) << 4, j0 = (w & 7) << 4;
      _Float16* hdst = hA + (size_t)(2 + d) * Bb * Hh;
#pragma unroll
      for (int r = 0; r < 8; r++)
        hdst[(size_t)(m0 + r + moff) * Hh + j0 + ncol] = (_Float16)hnew[i][r];
    }
    __syncthreads();

    // ---------------- logits (B x 48) ----------------
    for (int tile = wave; tile < 48; tile += 32) {
      const int m0 = (tile / 3) * 16;
      const int n0 = (tile % 3) * 16;
      v8f acc = vzero8();
#pragma unroll
      for (int kt = 0; kt < 8; ++kt) {
        const _Float16* src = (kt < 4) ? (hA + (size_t)2 * Bb * Hh) : (hA + (size_t)3 * Bb * Hh);
        const int k0 = (kt < 4) ? kt * 32 : (kt - 4) * 32;
        const v16h af = load_a_frag(src, Hh, m0, k0);
        acc = wmma16(af, load_b_frag(linf, kt, tile % 3, 3), acc);
      }
      const int n = n0 + ncol;
      const float lb = (n < Vv) ? lin_b[n] : -3.0e38f;  // pad cols never win argmax
#pragma unroll
      for (int r = 0; r < 8; r++) {
        const int m = m0 + r + moff;
        logits[(size_t)m * 48 + n] = acc[r] + lb;
      }
    }
    __syncthreads();

    // ---------------- argmax + blank flag ----------------
    if (threadIdx.x < Bb) {
      const int b = threadIdx.x;
      const float* lrow = logits + (size_t)b * 48;
      int best = 0;
      float bv = lrow[0];
      for (int v = 1; v < Vv; ++v) {
        const float lv = lrow[v];
        if (lv > bv) { bv = lv; best = v; }
      }
      const int f2 = flg[b] | (best == 0 ? 1 : 0);  // BLANK_IDX = 0
      flg[b] = f2;
      yv[b] = (float)best;
    }
    __syncthreads();

    // ---------------- output write (B,T,V) ----------------
    for (int e = threadIdx.x; e < Bb * Vv; e += blockDim.x) {
      const int b = e / Vv;
      const int v = e % Vv;
      const int f2 = flg[b];
      const float lv = logits[(size_t)b * 48 + v];
      out[((size_t)b * Tt + t) * Vv + v] =
          f2 ? ((v == 3) ? logits[(size_t)b * 48 + 3] : 0.f) : lv;  // KEEP_IDX = 3
    }
    __syncthreads();
  }
}

// ---------------------------------------------------------------------------
extern "C" void kernel_launch(void* const* d_in, const int* in_sizes, int n_in,
                              void* d_out, int out_size, void* d_ws, size_t ws_size,
                              hipStream_t stream) {
  (void)in_sizes; (void)n_in; (void)out_size; (void)ws_size;
  const int* x = (const int*)d_in[0];
  const float* h0 = (const float*)d_in[2];
  const float* c0 = (const float*)d_in[3];
  const float* enc_Wih0 = (const float*)d_in[4];
  const float* enc_Whh0 = (const float*)d_in[5];
  const float* enc_b0 = (const float*)d_in[6];
  const float* enc_Wih1 = (const float*)d_in[7];
  const float* enc_Whh1 = (const float*)d_in[8];
  const float* enc_b1 = (const float*)d_in[9];
  const float* dec_Wih0 = (const float*)d_in[10];
  const float* dec_Whh0 = (const float*)d_in[11];
  const float* dec_b0 = (const float*)d_in[12];
  const float* dec_Wih1 = (const float*)d_in[13];
  const float* dec_Whh1 = (const float*)d_in[14];
  const float* dec_b1 = (const float*)d_in[15];
  const float* lin_W = (const float*)d_in[16];
  const float* lin_b = (const float*)d_in[17];
  float* out = (float*)d_out;

  char* ws = (char*)d_ws;
  const size_t CUR_BYTES = (size_t)Ss * Bb * (2 * Hh) * sizeof(_Float16);  // 65.5 MB
  _Float16* cur_hi = (_Float16*)(ws);
  _Float16* cur_lo = (_Float16*)(ws + CUR_BYTES);
  float* h_fin = (float*)(ws + 2 * CUR_BYTES);
  float* c_fin = (float*)(ws + 2 * CUR_BYTES + (size_t)4 * Bb * Hh * sizeof(float));
  _Float16* frags = (_Float16*)(ws + 2 * CUR_BYTES + (size_t)8 * Bb * Hh * sizeof(float));

  const size_t WHH_FR = (size_t)4 * 32 * 512;  // halfs per Whh fragment set
  const size_t WIH_FR = (size_t)8 * 32 * 512;  // halfs per K=256 fragment set
  _Float16* f_enc_whh0 = frags;
  _Float16* f_enc_whh1 = f_enc_whh0 + 2 * WHH_FR;
  _Float16* f_enc_wih1 = f_enc_whh1 + 2 * WHH_FR;
  _Float16* f_dec_whh0 = f_enc_wih1 + 2 * WIH_FR;
  _Float16* f_dec_whh1 = f_dec_whh0 + 2 * WHH_FR;
  _Float16* f_dec_wih1 = f_dec_whh1 + 2 * WHH_FR;
  _Float16* f_lin = f_dec_wih1 + 2 * WIH_FR;

  dim3 pg(256), pb(256);
  for (int d = 0; d < 2; ++d) {
    plm_pack_frags<<<pg, pb, 0, stream>>>(enc_Whh0 + (size_t)d * Gg * Hh, f_enc_whh0 + d * WHH_FR, Hh, Gg, Gg);
    plm_pack_frags<<<pg, pb, 0, stream>>>(enc_Whh1 + (size_t)d * Gg * Hh, f_enc_whh1 + d * WHH_FR, Hh, Gg, Gg);
    plm_pack_frags<<<pg, pb, 0, stream>>>(enc_Wih1 + (size_t)d * Gg * (2 * Hh), f_enc_wih1 + d * WIH_FR, 2 * Hh, Gg, Gg);
    plm_pack_frags<<<pg, pb, 0, stream>>>(dec_Whh0 + (size_t)d * Gg * Hh, f_dec_whh0 + d * WHH_FR, Hh, Gg, Gg);
    plm_pack_frags<<<pg, pb, 0, stream>>>(dec_Whh1 + (size_t)d * Gg * Hh, f_dec_whh1 + d * WHH_FR, Hh, Gg, Gg);
    plm_pack_frags<<<pg, pb, 0, stream>>>(dec_Wih1 + (size_t)d * Gg * (2 * Hh), f_dec_wih1 + d * WIH_FR, 2 * Hh, Gg, Gg);
  }
  plm_pack_frags<<<pg, pb, 0, stream>>>(lin_W, f_lin, 2 * Hh, 48, Vv);

  const int enc_smem = (4 * 32 * 512 + 2 * Bb * Hh) * (int)sizeof(_Float16);  // 256 KB
  (void)hipFuncSetAttribute((const void*)plm_enc_l0, hipFuncAttributeMaxDynamicSharedMemorySize, enc_smem);
  (void)hipFuncSetAttribute((const void*)plm_enc_l1, hipFuncAttributeMaxDynamicSharedMemorySize, enc_smem);

  plm_enc_l0<<<2, 1024, enc_smem, stream>>>(x, h0, c0, enc_Wih0, enc_b0, f_enc_whh0,
                                            cur_hi, cur_lo, h_fin, c_fin);
  plm_enc_l1<<<2, 1024, enc_smem, stream>>>(h0, c0, enc_b1, f_enc_wih1, f_enc_whh1,
                                            cur_hi, cur_lo, h_fin, c_fin);

  const int dec_smem = 4 * Bb * Hh * 2 + Bb * 48 * 4 + Bb * 4 + Bb * 4;  // ~306 KB
  (void)hipFuncSetAttribute((const void*)plm_decoder, hipFuncAttributeMaxDynamicSharedMemorySize, dec_smem);
  plm_decoder<<<1, 1024, dec_smem, stream>>>(h_fin, c_fin, dec_Wih0, dec_b0, dec_b1,
                                             f_dec_whh0, f_dec_whh1, f_dec_wih1, f_lin,
                                             lin_b, out);
}
